// MoELayer_85272280695328
// MI455X (gfx1250) — compile-verified, matrix operand order checked by reading, not run on hardware
//
#include <hip/hip_runtime.h>

// ---------------------------------------------------------------------------
// MoE (top-2, SwiGLU experts) for MI455X / gfx1250, wave32 + WMMA bf16.
//   B=4 S=2048 H=1024 E=8 I=4096, T=8192 tokens.
// gate+gather tokens per expert -> bf16 convert/transpose weights once ->
// fused SwiGLU FFN with V_WMMA_F32_16X16X32_BF16 (f32 accum), x-tile staged
// into LDS with GLOBAL_LOAD_ASYNC_TO_LDS_B128 (ASYNCcnt), combine with
// global_atomic_add_f32.
// ---------------------------------------------------------------------------

typedef __bf16 bf16;
typedef __attribute__((ext_vector_type(16))) __bf16 v16bf;
typedef __attribute__((ext_vector_type(8)))  __bf16 v8bf;
typedef __attribute__((ext_vector_type(8)))  float  v8f;

constexpr int kB = 4, kS = 2048, kH = 1024, kE = 8, kI = 4096;
constexpr int kT  = kB * kS;   // 8192 tokens
constexpr int kM  = 32;        // tokens per FFN block
constexpr int kIC = 64;        // intermediate (I) chunk per iteration

// ---- workspace layout (bytes) ---------------------------------------------
constexpr size_t W_ELE   = (size_t)kE * kH * kI;          // 33.5M elements
constexpr size_t OFF_W1T = 0;                             // [E][I][H] bf16
constexpr size_t OFF_W3T = OFF_W1T + W_ELE * 2;           // [E][I][H] bf16
constexpr size_t OFF_W2T = OFF_W3T + W_ELE * 2;           // [E][H][I] bf16
constexpr size_t OFF_XBF = OFF_W2T + W_ELE * 2;           // [T][H] bf16
constexpr size_t OFF_CNT = OFF_XBF + (size_t)kT * kH * 2; // [E] int (padded)
constexpr size_t OFF_PTK = OFF_CNT + 256;                 // [E][T] int
constexpr size_t OFF_PWT = OFF_PTK + (size_t)kE * kT * 4; // [E][T] float
constexpr size_t WS_NEED = OFF_PWT + (size_t)kE * kT * 4; // ~218.6 MB

// ---------------------------------------------------------------------------
__device__ inline v8f wmma_bf16(v16bf a, v16bf b, v8f c) {
  // D = A(16x32) * B(32x16) + C, f32 accumulate
  return __builtin_amdgcn_wmma_f32_16x16x32_bf16(false, a, false, b,
                                                 (short)0, c, false, false);
}

// Build a 16-bit WMMA A/B fragment from a row-major row pointer.
// Layout (ISA 7.12.2): lanes 0-15: halves[0..7]=K(kk..kk+7), [8..15]=K(kk+16..+23)
//                      lanes 16-31: shift K by +8.  h8 = (lane&16)?8:0.
__device__ inline v16bf ldfrag(const bf16* __restrict__ row, int kk, int h8) {
  v8bf lo = *(const v8bf*)(row + kk + h8);
  v8bf hi = *(const v8bf*)(row + kk + 16 + h8);
  v16bf f;
#pragma unroll
  for (int j = 0; j < 8; ++j) { f[j] = lo[j]; f[8 + j] = hi[j]; }
  return f;
}

// Async 16B global->LDS copy, per lane (ISA §10.7 / §15.18.3 op 98).
// Low 32 bits of a generic pointer to LDS are the LDS byte address
// (ISA §10.2).
__device__ inline void async_g2l_b128(void* lds_dst, const void* gsrc) {
  unsigned lds = (unsigned)(size_t)lds_dst;
  asm volatile("global_load_async_to_lds_b128 %0, %1, off"
               :: "v"(lds), "v"(gsrc) : "memory");
}
__device__ inline void wait_asynccnt0() {
  asm volatile("s_wait_asynccnt 0x0" ::: "memory");
}

// ---------------------------------------------------------------------------
// 0) zero output + expert counters
__global__ void moe_zero(float* __restrict__ out, int n, int* __restrict__ cnt) {
  for (int i = blockIdx.x * blockDim.x + threadIdx.x; i < n;
       i += gridDim.x * blockDim.x)
    out[i] = 0.0f;
  if (blockIdx.x == 0 && threadIdx.x < kE) cnt[threadIdx.x] = 0;
}

// 1) gating: one wave per token; softmax -> top2 -> renormalize -> gather lists
__global__ __launch_bounds__(256) void moe_gate(
    const float* __restrict__ x, const float* __restrict__ gw,
    int* __restrict__ cnt, int* __restrict__ ptok, float* __restrict__ pwt) {
  int lane = threadIdx.x & 31;
  int wv   = threadIdx.x >> 5;
  int t    = blockIdx.x * 8 + wv;            // 1024 blocks * 8 waves = 8192
  const float* xr = x + (size_t)t * kH;
  float acc[kE];
#pragma unroll
  for (int e = 0; e < kE; ++e) acc[e] = 0.0f;
  for (int h = lane; h < kH; h += 32) {
    float xv = xr[h];
#pragma unroll
    for (int e = 0; e < kE; ++e) acc[e] += xv * gw[h * kE + e];
  }
#pragma unroll
  for (int e = 0; e < kE; ++e)
#pragma unroll
    for (int off = 16; off > 0; off >>= 1)
      acc[e] += __shfl_xor(acc[e], off, 32);
  if (lane == 0) {
    float m = acc[0];
#pragma unroll
    for (int e = 1; e < kE; ++e) m = fmaxf(m, acc[e]);
    float p[kE];
#pragma unroll
    for (int e = 0; e < kE; ++e) p[e] = __expf(acc[e] - m);
    int   i1 = 0; float v1 = p[0];
#pragma unroll
    for (int e = 1; e < kE; ++e) if (p[e] > v1) { v1 = p[e]; i1 = e; }
    int   i2 = (i1 == 0) ? 1 : 0; float v2 = p[i2];
#pragma unroll
    for (int e = 0; e < kE; ++e)
      if (e != i1 && p[e] > v2) { v2 = p[e]; i2 = e; }
    float inv = __builtin_amdgcn_rcpf(v1 + v2);
    int p1 = atomicAdd(&cnt[i1], 1);
    ptok[i1 * kT + p1] = t;  pwt[i1 * kT + p1] = v1 * inv;
    int p2 = atomicAdd(&cnt[i2], 1);
    ptok[i2 * kT + p2] = t;  pwt[i2 * kT + p2] = v2 * inv;
  }
}

// 2) x -> bf16
__global__ void moe_cvt_x(const float* __restrict__ src, bf16* __restrict__ dst,
                          int n) {
  for (int i = blockIdx.x * blockDim.x + threadIdx.x; i < n;
       i += gridDim.x * blockDim.x)
    dst[i] = (bf16)src[i];
}

// 3) fp32 [E][R][C] -> bf16 [E][C][R]  (LDS-tiled transpose+convert)
__global__ __launch_bounds__(256) void moe_tcvt(const float* __restrict__ src,
                                                bf16* __restrict__ dst,
                                                int R, int C) {
  __shared__ float tile[32][33];
  int e  = blockIdx.z;
  int c0 = blockIdx.x * 32, r0 = blockIdx.y * 32;
  int tx = threadIdx.x & 31, ty = threadIdx.x >> 5;   // ty in 0..7
  const float* s = src + ((size_t)e * R + r0) * C + c0;
#pragma unroll
  for (int it = 0; it < 4; ++it) {
    int r = ty + it * 8;
    tile[r][tx] = s[(size_t)r * C + tx];
  }
  __syncthreads();
  bf16* d = dst + ((size_t)e * C + c0) * R + r0;
#pragma unroll
  for (int it = 0; it < 4; ++it) {
    int c = ty + it * 8;
    d[(size_t)c * R + tx] = (bf16)tile[tx][c];
  }
}

// 4) fused SwiGLU expert FFN + weighted scatter-add combine
//    block = 32 gathered tokens of one expert, 8 waves (256 threads)
__global__ __launch_bounds__(256) void moe_ffn(
    const bf16* __restrict__ xbf,   // [T][H]
    const bf16* __restrict__ w1t,   // [E][I][H]
    const bf16* __restrict__ w3t,   // [E][I][H]
    const bf16* __restrict__ w2t,   // [E][H][I]
    const int*  __restrict__ cnt,   // [E]
    const int*  __restrict__ ptok,  // [E][T]
    const float* __restrict__ pwt,  // [E][T]
    float* __restrict__ out)        // [T][H]
{
  int e  = blockIdx.y;
  int n  = cnt[e];
  int m0 = blockIdx.x * kM;
  if (m0 >= n) return;                       // block-uniform: EXEC stays full

  __shared__ bf16  xs[kM][kH];               // 64 KB activation tile (bf16)
  __shared__ bf16  hs[kM][kIC];              // 4 KB SwiGLU tile
  __shared__ int   tok_s[kM];
  __shared__ float rw_s[kM];

  int tid  = threadIdx.x;
  int w    = tid >> 5;
  int lane = tid & 31;
  int ln   = lane & 15;
  int h8   = (lane & 16) ? 8 : 0;
  int rt   = w >> 2;                         // row-tile 0/1 (16 rows each)
  int nt   = w & 3;                          // GEMM1 column tile within chunk

  if (tid < kM) {
    int p = m0 + tid;
    tok_s[tid] = (p < n) ? ptok[e * kT + p] : 0;
    rw_s[tid]  = (p < n) ? pwt[e * kT + p] : 0.0f;
  }
  __syncthreads();

  // stage 32 gathered x rows into LDS: async global->LDS b128, no VGPR bounce
  {
    int row = tid >> 3;
    int cb  = (tid & 7) * 128;
    const bf16* src = xbf + (size_t)tok_s[row] * kH + cb;
#pragma unroll
    for (int j = 0; j < 16; ++j)
      async_g2l_b128(&xs[row][cb + j * 8], src + j * 8);
  }
  wait_asynccnt0();
  __syncthreads();

  const bf16* w1e = w1t + (size_t)e * kI * kH;
  const bf16* w3e = w3t + (size_t)e * kI * kH;
  const bf16* w2e = w2t + (size_t)e * kH * kI;

  const bf16* xrow  = &xs[rt * 16 + ln][0];
  const bf16* h2row = &hs[rt * 16 + ln][0];

  v8f acc[16];                               // y tile: rows rt*16.., cols (w&3)*256..
#pragma unroll
  for (int j = 0; j < 16; ++j) acc[j] = (v8f){};

  for (int i0 = 0; i0 < kI; i0 += kIC) {
    // ---------------- GEMM1: a1 = x@w1e, a3 = x@w3e (K = H = 1024) --------
    const bf16* w1row = w1e + (size_t)(i0 + nt * 16 + ln) * kH;
    const bf16* w3row = w3e + (size_t)(i0 + nt * 16 + ln) * kH;
    if (i0 + kIC < kI) {                     // stream next chunk's weights
      __builtin_prefetch(w1row + (size_t)kIC * kH, 0, 3);
      __builtin_prefetch(w3row + (size_t)kIC * kH, 0, 3);
    }
    v8f a1 = (v8f){}, a3 = (v8f){};
#pragma unroll 4
    for (int kk = 0; kk < kH; kk += 32) {
      v16bf af = ldfrag(xrow,  kk, h8);
      v16bf b1 = ldfrag(w1row, kk, h8);
      v16bf b3 = ldfrag(w3row, kk, h8);
      a1 = wmma_bf16(af, b1, a1);
      a3 = wmma_bf16(af, b3, a3);
    }

    // make sure previous chunk's hs reads are done before overwriting
    __syncthreads();

    // SwiGLU + routing weight, write bf16 h-tile to LDS
    // silu(z)*a3*w with fast reciprocal (output is bf16 anyway)
#pragma unroll
    for (int r = 0; r < 8; ++r) {
      int row = rt * 16 + r + h8;            // C layout: VGPR r -> M = r(+8)
      float z  = a1[r];
      float hv = z * __builtin_amdgcn_rcpf(1.0f + __expf(-z)) * a3[r] * rw_s[row];
      hs[row][nt * 16 + ln] = (bf16)hv;
    }
    __syncthreads();

    // ---------------- GEMM2: y += h @ w2e  (K = kIC = 64) -----------------
    v16bf a2k0 = ldfrag(h2row, 0,  h8);
    v16bf a2k1 = ldfrag(h2row, 32, h8);
    int cb = (w & 3) * 256 + ln;
#pragma unroll
    for (int j = 0; j < 16; ++j) {
      const bf16* w2row = w2e + (size_t)(cb + j * 16) * kI;
      acc[j] = wmma_bf16(a2k0, ldfrag(w2row, i0,      h8), acc[j]);
      acc[j] = wmma_bf16(a2k1, ldfrag(w2row, i0 + 32, h8), acc[j]);
    }
  }

  // weighted scatter-add combine (top-2 => tokens receive 2 contributions)
#pragma unroll
  for (int j = 0; j < 16; ++j) {
    int col = (w & 3) * 256 + j * 16 + ln;
#pragma unroll
    for (int r = 0; r < 8; ++r) {
      int row = rt * 16 + r + h8;
      atomicAdd(&out[(size_t)tok_s[row] * kH + col], acc[j][r]);
    }
  }
}

// ---------------------------------------------------------------------------
extern "C" void kernel_launch(void* const* d_in, const int* in_sizes, int n_in,
                              void* d_out, int out_size, void* d_ws,
                              size_t ws_size, hipStream_t stream) {
  (void)in_sizes; (void)n_in; (void)out_size;
  const float* x  = (const float*)d_in[0];   // [B,S,H]
  const float* gw = (const float*)d_in[1];   // [H,E]
  const float* w1 = (const float*)d_in[2];   // [E,H,I]
  const float* w3 = (const float*)d_in[3];   // [E,H,I]
  const float* w2 = (const float*)d_in[4];   // [E,I,H]
  float* out = (float*)d_out;                // [B,S,H] fp32

  if (ws_size < WS_NEED) return;             // need ~219 MB scratch
  char* ws = (char*)d_ws;
  bf16*  w1t  = (bf16*)(ws + OFF_W1T);
  bf16*  w3t  = (bf16*)(ws + OFF_W3T);
  bf16*  w2t  = (bf16*)(ws + OFF_W2T);
  bf16*  xbf  = (bf16*)(ws + OFF_XBF);
  int*   cnt  = (int*)(ws + OFF_CNT);
  int*   ptk  = (int*)(ws + OFF_PTK);
  float* pwt  = (float*)(ws + OFF_PWT);

  const int nOut = kT * kH;                  // 8.39M floats

  moe_zero<<<4096, 256, 0, stream>>>(out, nOut, cnt);
  moe_gate<<<kT / 8, 256, 0, stream>>>(x, gw, cnt, ptk, pwt);
  moe_cvt_x<<<8192, 256, 0, stream>>>(x, xbf, nOut);
  // w1,w3: [E,H,I] -> [E,I,H];  w2: [E,I,H] -> [E,H,I]
  moe_tcvt<<<dim3(kI / 32, kH / 32, kE), 256, 0, stream>>>(w1, w1t, kH, kI);
  moe_tcvt<<<dim3(kI / 32, kH / 32, kE), 256, 0, stream>>>(w3, w3t, kH, kI);
  moe_tcvt<<<dim3(kH / 32, kI / 32, kE), 256, 0, stream>>>(w2, w2t, kI, kH);
  moe_ffn<<<dim3(kT / kM, kE), 256, 0, stream>>>(xbf, w1t, w3t, w2t,
                                                 cnt, ptk, pwt, out);
}
